// PointNetPp_41412074668138
// MI455X (gfx1250) — compile-verified, matrix operand order checked by reading, not run on hardware
//
#include <hip/hip_runtime.h>
#include <hip/hip_bf16.h>

typedef _Float16 v16h __attribute__((ext_vector_type(16)));
typedef _Float16 h8   __attribute__((ext_vector_type(8)));
typedef float    v8f  __attribute__((ext_vector_type(8)));

// ---------------------------------------------------------------------------
// prep: feat (B,6,N) f32 -> xyz (B,N,3) f32 and l0_points (B,N,6) f16
// ---------------------------------------------------------------------------
__global__ void prep_inputs(const float* __restrict__ feat, float* __restrict__ xyz,
                            _Float16* __restrict__ l0p, int N) {
    int t = blockIdx.x * blockDim.x + threadIdx.x;
    int total = 2 * N;
    if (t >= total) return;
    int b = t / N, nn = t % N;
    for (int c = 0; c < 6; ++c) {
        float v = feat[((size_t)b * 6 + c) * N + nn];
        l0p[((size_t)b * N + nn) * 6 + c] = (_Float16)v;
        if (c < 3) xyz[((size_t)b * N + nn) * 3 + c] = v;
    }
}

// ---------------------------------------------------------------------------
// farthest point sampling: one workgroup per batch, sequential npoint iters.
// Emits old 'far' each iter (matches lax.scan semantics), argmax ties -> lowest idx.
// ---------------------------------------------------------------------------
#define FPS_T 512
__global__ __launch_bounds__(FPS_T)
void fps_kernel(const float* __restrict__ xyz, float* __restrict__ dist,
                int* __restrict__ outIdx, int n, int npoint) {
    int b = blockIdx.x;
    int tid = threadIdx.x;
    __shared__ float sv[FPS_T];
    __shared__ int   si[FPS_T];
    __shared__ float cen[3];
    __shared__ int   sFar;
    const float* X = xyz + (size_t)b * n * 3;
    float* D = dist + (size_t)b * n;
    for (int i = tid; i < n; i += FPS_T) D[i] = 1e10f;
    if (tid == 0) sFar = 0;
    __syncthreads();
    for (int it = 0; it < npoint; ++it) {
        int far = sFar;
        if (tid == 0) {
            outIdx[b * npoint + it] = far;
            cen[0] = X[(size_t)far * 3 + 0];
            cen[1] = X[(size_t)far * 3 + 1];
            cen[2] = X[(size_t)far * 3 + 2];
        }
        __syncthreads();
        float cx = cen[0], cy = cen[1], cz = cen[2];
        float bv = -1.0f; int bi = 0x7fffffff;
        for (int i = tid; i < n; i += FPS_T) {
            float dx = X[(size_t)i * 3 + 0] - cx;
            float dy = X[(size_t)i * 3 + 1] - cy;
            float dz = X[(size_t)i * 3 + 2] - cz;
            float d = dx * dx + dy * dy + dz * dz;
            float old = D[i];
            d = d < old ? d : old;
            D[i] = d;
            if (d > bv) { bv = d; bi = i; }  // ascending i => first max kept
        }
        sv[tid] = bv; si[tid] = bi;
        __syncthreads();
        for (int s = FPS_T / 2; s > 0; s >>= 1) {
            if (tid < s) {
                float v2 = sv[tid + s]; int i2 = si[tid + s];
                if (v2 > sv[tid] || (v2 == sv[tid] && i2 < si[tid])) { sv[tid] = v2; si[tid] = i2; }
            }
            __syncthreads();
        }
        if (tid == 0) sFar = si[0];
        __syncthreads();
    }
}

__global__ void gather_xyz(const float* __restrict__ xyz, const int* __restrict__ fidx,
                           float* __restrict__ nxyz, int n, int np) {
    int t = blockIdx.x * blockDim.x + threadIdx.x;
    if (t >= 2 * np) return;
    int b = t / np, p = t % np;
    int s = fidx[b * np + p];
    for (int c = 0; c < 3; ++c)
        nxyz[((size_t)b * np + p) * 3 + c] = xyz[((size_t)b * n + s) * 3 + c];
}

// ---------------------------------------------------------------------------
// ball query: first k indices (ascending) with d2 <= r2, padded with first.
// ---------------------------------------------------------------------------
__global__ void ball_query(const float* __restrict__ xyz, const float* __restrict__ nxyz,
                           int* __restrict__ idx, int n, int np, float r2, int k) {
    int t = blockIdx.x * blockDim.x + threadIdx.x;
    if (t >= 2 * np) return;
    int b = t / np, p = t % np;
    const float* q = nxyz + ((size_t)b * np + p) * 3;
    float qx = q[0], qy = q[1], qz = q[2];
    int* o = idx + ((size_t)b * np + p) * k;
    int cnt = 0;
    for (int j = 0; j < n && cnt < k; ++j) {
        const float* s = xyz + ((size_t)b * n + j) * 3;
        float dx = s[0] - qx, dy = s[1] - qy, dz = s[2] - qz;
        if (dx * dx + dy * dy + dz * dz <= r2) { o[cnt] = j; ++cnt; }
    }
    int first = (cnt > 0) ? o[0] : 0;
    for (int j = cnt; j < k; ++j) o[j] = first;
}

// ---------------------------------------------------------------------------
// grouped A builder: A[(b*np+p)*k + j, :] = [ points[idx] (C) | xyz[idx]-new_xyz (3) | 0 pad ]
// ---------------------------------------------------------------------------
__global__ void build_group_A(const float* __restrict__ xyz, const float* __restrict__ nxyz,
                              const _Float16* __restrict__ pts, const int* __restrict__ idx,
                              _Float16* __restrict__ A, int n, int np, int k, int C, int Kp, int total) {
    int t = blockIdx.x * blockDim.x + threadIdx.x;
    if (t >= total) return;
    int c = t % Kp;
    int m = t / Kp;       // (b*np + p)*k + j
    int bp = m / k;       // b*np + p
    int src = idx[(size_t)m];
    int b = bp / np;
    _Float16 v = (_Float16)0.0f;
    if (c < C) {
        v = pts[((size_t)b * n + src) * C + c];
    } else if (c < C + 3) {
        int cc = c - C;
        v = (_Float16)(xyz[((size_t)b * n + src) * 3 + cc] - nxyz[(size_t)bp * 3 + cc]);
    }
    A[(size_t)m * Kp + c] = v;
}

// ---------------------------------------------------------------------------
// 3-NN + inverse-distance weights for feature propagation
// ---------------------------------------------------------------------------
__global__ void knn3_kernel(const float* __restrict__ x1, const float* __restrict__ x2,
                            int* __restrict__ idx3, float* __restrict__ w3, int n1, int n2) {
    int t = blockIdx.x * blockDim.x + threadIdx.x;
    if (t >= 2 * n1) return;
    int b = t / n1, i = t % n1;
    const float* p = x1 + ((size_t)b * n1 + i) * 3;
    float px = p[0], py = p[1], pz = p[2];
    float d0 = 3e38f, d1 = 3e38f, d2v = 3e38f;
    int i0 = 0, i1 = 0, i2 = 0;
    for (int j = 0; j < n2; ++j) {
        const float* q = x2 + ((size_t)b * n2 + j) * 3;
        float dx = q[0] - px, dy = q[1] - py, dz = q[2] - pz;
        float d = dx * dx + dy * dy + dz * dz;
        if (d < d0)      { d2v = d1; i2 = i1; d1 = d0; i1 = i0; d0 = d; i0 = j; }
        else if (d < d1) { d2v = d1; i2 = i1; d1 = d;  i1 = j; }
        else if (d < d2v){ d2v = d;  i2 = j; }
    }
    float w0 = 1.f / (d0 + 1e-8f), w1 = 1.f / (d1 + 1e-8f), w2 = 1.f / (d2v + 1e-8f);
    float ws = w0 + w1 + w2;
    size_t o = ((size_t)b * n1 + i) * 3;
    idx3[o] = i0; idx3[o + 1] = i1; idx3[o + 2] = i2;
    w3[o] = w0 / ws; w3[o + 1] = w1 / ws; w3[o + 2] = w2 / ws;
}

// ---------------------------------------------------------------------------
// fp A builder: A[b*n1+i, :] = [ points1 (C1) | sum_t w[t]*points2[idx[t]] (C2) | 0 pad ]
// ---------------------------------------------------------------------------
__global__ void build_fp_A(const _Float16* __restrict__ p1, const _Float16* __restrict__ p2,
                           const int* __restrict__ idx3, const float* __restrict__ w3,
                           _Float16* __restrict__ A, int n1, int n2, int C1, int C2, int Kp, int total) {
    int t = blockIdx.x * blockDim.x + threadIdx.x;
    if (t >= total) return;
    int c = t % Kp;
    int m = t / Kp;   // b*n1 + i
    int b = m / n1;
    _Float16 v = (_Float16)0.0f;
    if (c < C1) {
        v = p1[(size_t)m * C1 + c];
    } else if (c < C1 + C2) {
        int cc = c - C1;
        size_t o = (size_t)m * 3;
        float acc = 0.f;
        for (int tt = 0; tt < 3; ++tt)
            acc += w3[o + tt] * (float)p2[((size_t)b * n2 + idx3[o + tt]) * C2 + cc];
        v = (_Float16)acc;
    }
    A[(size_t)m * Kp + c] = v;
}

// ---------------------------------------------------------------------------
// weight staging: w f32 (O,K) -> f16 (Opad, Kpad), zero padded
// ---------------------------------------------------------------------------
__global__ void conv_w(const float* __restrict__ w, _Float16* __restrict__ wp,
                       int O, int K, int Op, int Kp) {
    int t = blockIdx.x * blockDim.x + threadIdx.x;
    int total = Op * Kp;
    if (t >= total) return;
    int o = t / Kp, kk = t % Kp;
    float v = (o < O && kk < K) ? w[(size_t)o * K + kk] : 0.f;
    wp[t] = (_Float16)v;
}

// ---------------------------------------------------------------------------
// WMMA GEMM: Y(M,Opad) f16 = A(M,Kp) f16 * W(Opad,Kp)^T, f32 accumulate.
// Block = 4 waves; wave computes 16x64 (one A frag reused across 4 B frags).
// M % 64 == 0, Kp % 32 == 0, Opad % 64 == 0 (guaranteed by host).
// ---------------------------------------------------------------------------
__global__ __launch_bounds__(128)
void gemm_f16_wmma(const _Float16* __restrict__ A, const _Float16* __restrict__ W,
                   _Float16* __restrict__ Y, int M, int Kp, int Opad) {
    int wave = threadIdx.x >> 5;
    int lane = threadIdx.x & 31;
    int lo = lane & 15, hi = lane >> 4;
    int m0 = blockIdx.x * 64 + wave * 16;
    int o0 = blockIdx.y * 64;
    const _Float16* arow = A + (size_t)(m0 + lo) * Kp;
    v8f acc0 = {}, acc1 = {}, acc2 = {}, acc3 = {};
    for (int k0 = 0; k0 < Kp; k0 += 32) {
        v16h af;
        ((h8*)&af)[0] = *(const h8*)(arow + k0 + hi * 8);
        ((h8*)&af)[1] = *(const h8*)(arow + k0 + 16 + hi * 8);
        const _Float16* wr0 = W + (size_t)(o0 + 0  + lo) * Kp + k0 + hi * 8;
        const _Float16* wr1 = W + (size_t)(o0 + 16 + lo) * Kp + k0 + hi * 8;
        const _Float16* wr2 = W + (size_t)(o0 + 32 + lo) * Kp + k0 + hi * 8;
        const _Float16* wr3 = W + (size_t)(o0 + 48 + lo) * Kp + k0 + hi * 8;
        v16h b0, b1, b2, b3;
        ((h8*)&b0)[0] = *(const h8*)(wr0);      ((h8*)&b0)[1] = *(const h8*)(wr0 + 16);
        ((h8*)&b1)[0] = *(const h8*)(wr1);      ((h8*)&b1)[1] = *(const h8*)(wr1 + 16);
        ((h8*)&b2)[0] = *(const h8*)(wr2);      ((h8*)&b2)[1] = *(const h8*)(wr2 + 16);
        ((h8*)&b3)[0] = *(const h8*)(wr3);      ((h8*)&b3)[1] = *(const h8*)(wr3 + 16);
        acc0 = __builtin_amdgcn_wmma_f32_16x16x32_f16(false, af, false, b0, (short)0, acc0, false, false);
        acc1 = __builtin_amdgcn_wmma_f32_16x16x32_f16(false, af, false, b1, (short)0, acc1, false, false);
        acc2 = __builtin_amdgcn_wmma_f32_16x16x32_f16(false, af, false, b2, (short)0, acc2, false, false);
        acc3 = __builtin_amdgcn_wmma_f32_16x16x32_f16(false, af, false, b3, (short)0, acc3, false, false);
    }
    v8f accs[4] = {acc0, acc1, acc2, acc3};
#pragma unroll
    for (int t = 0; t < 4; ++t) {
        int col = o0 + t * 16 + lo;
#pragma unroll
        for (int j = 0; j < 8; ++j) {
            int row = m0 + j + 8 * hi;
            Y[(size_t)row * Opad + col] = (_Float16)accs[t][j];
        }
    }
}

// ---------------------------------------------------------------------------
// BN stats: one block per channel, deterministic tree reduction.
// mean = sum/M, var = sumsq/M - mean^2 (biased, like jnp.var)
// ---------------------------------------------------------------------------
__global__ __launch_bounds__(256)
void bn_stats(const _Float16* __restrict__ Y, int M, int Op,
              float* __restrict__ mean, float* __restrict__ var) {
    int c = blockIdx.x;
    int tid = threadIdx.x;
    __shared__ float ss[256], sq[256];
    float s = 0.f, q = 0.f;
    for (int r = tid; r < M; r += 256) {
        float v = (float)Y[(size_t)r * Op + c];
        s += v; q += v * v;
    }
    ss[tid] = s; sq[tid] = q;
    __syncthreads();
    for (int st = 128; st > 0; st >>= 1) {
        if (tid < st) { ss[tid] += ss[tid + st]; sq[tid] += sq[tid + st]; }
        __syncthreads();
    }
    if (tid == 0) {
        float m = ss[0] / (float)M;
        mean[c] = m;
        var[c] = sq[0] / (float)M - m * m;
    }
}

// normalize + relu -> f16 out with arbitrary out stride (zero pads c >= O)
__global__ void bn_norm_relu(const _Float16* __restrict__ Y, int M, int O, int Op,
                             const float* __restrict__ mean, const float* __restrict__ var,
                             const float* __restrict__ g, const float* __restrict__ beta,
                             _Float16* __restrict__ out, int outStride, int total) {
    int t = blockIdx.x * blockDim.x + threadIdx.x;
    if (t >= total) return;
    int c = t % outStride;
    int m = t / outStride;
    _Float16 r = (_Float16)0.0f;
    if (c < O) {
        float v = (float)Y[(size_t)m * Op + c];
        v = (v - mean[c]) * rsqrtf(var[c] + 1e-5f) * g[c] + beta[c];
        r = (_Float16)(v > 0.f ? v : 0.f);
    }
    out[(size_t)m * outStride + c] = r;
}

// fused BN + relu + max over group dim k -> concat destination (stride Cs, offset coff)
__global__ void bn_relu_maxpool(const _Float16* __restrict__ Y, int k, int O, int Op,
                                const float* __restrict__ mean, const float* __restrict__ var,
                                const float* __restrict__ g, const float* __restrict__ beta,
                                _Float16* __restrict__ out, int Cs, int coff, int total) {
    int t = blockIdx.x * blockDim.x + threadIdx.x;
    if (t >= total) return;
    int c = t % O;
    int bp = t / O;
    float sc = rsqrtf(var[c] + 1e-5f) * g[c];
    float mu = mean[c], bt = beta[c];
    float best = -3e38f;
    const _Float16* base = Y + (size_t)bp * k * Op + c;
    for (int j = 0; j < k; ++j) {
        float v = (float)base[(size_t)j * Op];
        v = (v - mu) * sc + bt;
        v = v > 0.f ? v : 0.f;
        if (v > best) best = v;
    }
    out[(size_t)bp * Cs + coff + c] = (_Float16)best;
}

// final 17x17 linear (+bias), writes transposed output (B,17,N) f32
__global__ void cls2_kernel(const _Float16* __restrict__ cbuf, const float* __restrict__ w2,
                            const float* __restrict__ b2, float* __restrict__ out, int N, int total) {
    int t = blockIdx.x * blockDim.x + threadIdx.x;
    if (t >= total) return;
    int nn = t % N;
    int rest = t / N;
    int o = rest % 17;
    int b = rest / 17;
    float acc = b2[o];
    const _Float16* cp = cbuf + ((size_t)b * N + nn) * 17;
    for (int c = 0; c < 17; ++c) acc += (float)cp[c] * w2[o * 17 + c];
    out[((size_t)b * 17 + o) * N + nn] = acc;
}

// ---------------------------------------------------------------------------
// host
// ---------------------------------------------------------------------------
static inline int aln(int x, int a) { return (x + a - 1) / a * a; }

extern "C" void kernel_launch(void* const* d_in, const int* in_sizes, int n_in,
                              void* d_out, int out_size, void* d_ws, size_t ws_size,
                              hipStream_t stream) {
    (void)out_size; (void)ws_size;
    const int N = 24000;

    // param base indices: disambiguate dict flatten order at runtime
    int i_cls1, i_cls2, i_fp1, i_fp2, i_fp3, i_sa1, i_sa2, i_sa3;
    if (n_in > 1 && in_sizes[1] == 17 * 128) {
        // sorted keys: cls1, cls2, fp1, fp2, fp3, sa1, sa2, sa3
        i_cls1 = 1; i_cls2 = 5; i_fp1 = 7; i_fp2 = 15; i_fp3 = 23;
        i_sa1 = 31; i_sa2 = 47; i_sa3 = 63;
    } else {
        // insertion order: sa1, sa2, sa3, fp3, fp2, fp1, cls1, cls2
        i_sa1 = 1; i_sa2 = 17; i_sa3 = 33; i_fp3 = 49; i_fp2 = 57; i_fp1 = 65;
        i_cls1 = 73; i_cls2 = 77;
    }
    auto P = [&](int idx) -> const float* { return (const float*)d_in[idx]; };
    const float* feat = P(0);

    // workspace carve
    char* wp = (char*)d_ws;
    auto carve = [&](size_t bytes) -> char* {
        char* r = wp; wp += (bytes + 255) & ~(size_t)255; return r;
    };
    float*    xyz0    = (float*)carve((size_t)2 * N * 3 * 4);
    _Float16* l0p     = (_Float16*)carve((size_t)2 * N * 6 * 2);
    float*    l1x     = (float*)carve((size_t)2 * 1024 * 3 * 4);
    float*    l2x     = (float*)carve((size_t)2 * 512 * 3 * 4);
    float*    l3x     = (float*)carve((size_t)2 * 256 * 3 * 4);
    _Float16* l1p     = (_Float16*)carve((size_t)2 * 1024 * 256 * 2);
    _Float16* l2p     = (_Float16*)carve((size_t)2 * 512 * 1024 * 2);
    _Float16* l3p     = (_Float16*)carve((size_t)2 * 256 * 2048 * 2);
    _Float16* l2pn    = (_Float16*)carve((size_t)2 * 512 * 1024 * 2);
    _Float16* l1pn    = (_Float16*)carve((size_t)2 * 1024 * 512 * 2);
    _Float16* l0pp    = (_Float16*)carve((size_t)2 * N * 128 * 2);
    int*      fpsIdx  = (int*)carve((size_t)2 * 1024 * 4);
    float*    distBuf = (float*)carve((size_t)2 * N * 4);
    int*      ballIdx = (int*)carve((size_t)2 * 1024 * 64 * 4);
    int*      idx3    = (int*)carve((size_t)2 * N * 3 * 4);
    float*    w3      = (float*)carve((size_t)2 * N * 3 * 4);
    _Float16* cbuf    = (_Float16*)carve((size_t)2 * N * 17 * 2);
    float*    meanB   = (float*)carve((size_t)1024 * 4);
    float*    varB    = (float*)carve((size_t)1024 * 4);
    _Float16* Wbuf    = (_Float16*)carve((size_t)1024 * 3072 * 2);        // max Opad*Kp (fp3 L1)
    _Float16* Abuf    = (_Float16*)carve((size_t)32768 * 1056 * 2);       // max M*Kp (sa3 br1 L1)
    _Float16* Ybuf    = (_Float16*)carve((size_t)32768 * 1024 * 2);       // max M*Opad

    // staged GEMM + batch stats (bias b is absorbed by BN -> skipped exactly)
    auto run_gemm_stats = [&](const _Float16* A, const float* w, int M, int K, int O) {
        int Kp = aln(K, 32), Op = aln(O, 64);
        int nW = Op * Kp;
        conv_w<<<(nW + 255) / 256, 256, 0, stream>>>(w, Wbuf, O, K, Op, Kp);
        gemm_f16_wmma<<<dim3(M / 64, Op / 64), 128, 0, stream>>>(A, Wbuf, Ybuf, M, Kp, Op);
        bn_stats<<<O, 256, 0, stream>>>(Ybuf, M, Op, meanB, varB);
    };

    // SA (set abstraction, MSG) stage
    auto sa_stage = [&](const float* xyzIn, int n, const _Float16* pts, int C,
                        int np, float r0, float r1, int k0, int k1, int base,
                        int h, int o, float* nxyz, _Float16* outp) {
        fps_kernel<<<2, FPS_T, 0, stream>>>(xyzIn, distBuf, fpsIdx, n, np);
        gather_xyz<<<(2 * np + 127) / 128, 128, 0, stream>>>(xyzIn, fpsIdx, nxyz, n, np);
        float rr[2] = {r0, r1};
        int   kk[2] = {k0, k1};
        int Cs = 2 * o;
        for (int br = 0; br < 2; ++br) {
            int k = kk[br];
            float r2 = rr[br] * rr[br];
            ball_query<<<(2 * np + 127) / 128, 128, 0, stream>>>(xyzIn, nxyz, ballIdx, n, np, r2, k);
            int M = 2 * np * k;
            int K1 = C + 3, Kp1 = aln(K1, 32);
            int tot1 = M * Kp1;
            build_group_A<<<(tot1 + 255) / 256, 256, 0, stream>>>(xyzIn, nxyz, pts, ballIdx,
                                                                  Abuf, n, np, k, C, Kp1, tot1);
            const float* w1 = P(base + 8 * br + 0);
            const float* g1 = P(base + 8 * br + 2);
            const float* e1 = P(base + 8 * br + 3);
            run_gemm_stats(Abuf, w1, M, K1, h);
            int Kp2 = aln(h, 32);
            int tot2 = M * Kp2;
            bn_norm_relu<<<(tot2 + 255) / 256, 256, 0, stream>>>(Ybuf, M, h, aln(h, 64),
                                                                 meanB, varB, g1, e1, Abuf, Kp2, tot2);
            const float* w2 = P(base + 8 * br + 4);
            const float* g2 = P(base + 8 * br + 6);
            const float* e2 = P(base + 8 * br + 7);
            run_gemm_stats(Abuf, w2, M, h, o);
            int tot3 = 2 * np * o;
            bn_relu_maxpool<<<(tot3 + 255) / 256, 256, 0, stream>>>(Ybuf, k, o, aln(o, 64),
                                                                    meanB, varB, g2, e2,
                                                                    outp, Cs, br * o, tot3);
        }
    };

    // FP (feature propagation) stage
    auto fp_stage = [&](const float* x1, int n1, const float* x2, int n2,
                        const _Float16* p1, int C1, const _Float16* p2, int C2,
                        int base, int h, int o, _Float16* outp) {
        knn3_kernel<<<(2 * n1 + 127) / 128, 128, 0, stream>>>(x1, x2, idx3, w3, n1, n2);
        int M = 2 * n1, K1 = C1 + C2, Kp1 = aln(K1, 32);
        int tot1 = M * Kp1;
        build_fp_A<<<(tot1 + 255) / 256, 256, 0, stream>>>(p1, p2, idx3, w3, Abuf,
                                                           n1, n2, C1, C2, Kp1, tot1);
        run_gemm_stats(Abuf, P(base + 0), M, K1, h);
        int Kp2 = aln(h, 32);
        int tot2 = M * Kp2;
        bn_norm_relu<<<(tot2 + 255) / 256, 256, 0, stream>>>(Ybuf, M, h, aln(h, 64),
                                                             meanB, varB, P(base + 2), P(base + 3),
                                                             Abuf, Kp2, tot2);
        run_gemm_stats(Abuf, P(base + 4), M, h, o);
        int tot3 = M * o;
        bn_norm_relu<<<(tot3 + 255) / 256, 256, 0, stream>>>(Ybuf, M, o, aln(o, 64),
                                                             meanB, varB, P(base + 6), P(base + 7),
                                                             outp, o, tot3);
    };

    // ---- pipeline ----
    prep_inputs<<<(2 * N + 255) / 256, 256, 0, stream>>>(feat, xyz0, l0p, N);

    sa_stage(xyz0, N,    l0p, 6,    1024, 0.025f, 0.05f, 32, 64, i_sa1, 128, 128,  l1x, l1p);
    sa_stage(l1x,  1024, l1p, 256,  512,  0.05f,  0.1f,  32, 64, i_sa2, 256, 512,  l2x, l2p);
    sa_stage(l2x,  512,  l2p, 1024, 256,  0.1f,   0.2f,  32, 64, i_sa3, 784, 1024, l3x, l3p);

    fp_stage(l2x, 512,  l3x, 256,  l2p, 1024, l3p,  2048, i_fp3, 1024, 1024, l2pn);
    fp_stage(l1x, 1024, l2x, 512,  l1p, 256,  l2pn, 1024, i_fp2, 512,  512,  l1pn);
    fp_stage(xyz0, N,   l1x, 1024, l0p, 6,    l1pn, 512,  i_fp1, 256,  128,  l0pp);

    // classifier head: conv_bn_relu(128 -> 17) then 17x17 linear + transpose
    run_gemm_stats(l0pp, P(i_cls1 + 0), 2 * N, 128, 17);
    int totc = 2 * N * 17;
    bn_norm_relu<<<(totc + 255) / 256, 256, 0, stream>>>(Ybuf, 2 * N, 17, 64,
                                                         meanB, varB, P(i_cls1 + 2), P(i_cls1 + 3),
                                                         cbuf, 17, totc);
    int toto = 2 * 17 * N;
    cls2_kernel<<<(toto + 255) / 256, 256, 0, stream>>>(cbuf, P(i_cls2 + 0), P(i_cls2 + 1),
                                                        (float*)d_out, N, toto);
}
